// Generator_22076131902158
// MI455X (gfx1250) — matrix operand order repaired and checked by reading, not compiled, not run on hardware
//
#include <hip/hip_runtime.h>

// ---------------------------------------------------------------------------
// GRU decoder, fully fused, bf16 WMMA (fp32 accumulate) on gfx1250 (wave32).
//   h0 = tanh(z @ fc_w^T + fc_b)
//   for t in 0..179:  gh = h @ w_hh^T + b_hh ; GRU gate update (gi == b_ih)
//                     out[:,t,:] = h @ out_w^T + out_b      (fused, no hs buffer)
// 64 blocks x 256 threads (8 wave32); each wave owns a 16-row M tile.
// w_hh: pre-converted to bf16 in d_ws, streamed L2 -> LDS two j-tiles at a
// time with GLOBAL_LOAD_ASYNC_TO_LDS_B128 (double-buffered pairs, ASYNCcnt),
// halving barrier count and doubling the prefetch window. Gates use hardware
// V_TANH_F32. fp32-streaming fallback when workspace unavailable.
// ---------------------------------------------------------------------------

#define SEQ_LEN 180
#define LATENT  128
#define HIDDEN  256
#define BATCH   8192

#define THREADS 256
#define ROWS_PER_BLOCK 128
#define NBLOCKS (BATCH / ROWS_PER_BLOCK)   // 64

// padded LDS row strides (elements) to avoid bank conflicts
#define HF_STRIDE 260   // floats  (256 + 4)
#define HB_STRIDE 264   // halves  (256 + 8)
#define BS_STRIDE 264   // halves
#define ZB_STRIDE 136   // halves  (128 + 8)

#define BS_HALVES  (3 * 16 * BS_STRIDE)                      // one j-tile, halves (12672)

// LDS byte layout
#define HF_BYTES   (ROWS_PER_BLOCK * HF_STRIDE * 4)          // 133120
#define HB_OFF     HF_BYTES                                  // 133120
#define HB_BYTES   (ROWS_PER_BLOCK * HB_STRIDE * 2)          // 67584
#define BS_OFF     (HB_OFF + HB_BYTES)                       // 200704
#define BS_BYTES   (4 * BS_HALVES * 2)                       // 101376 (2 pair-buffers)
#define ZB_OFF     (BS_OFF + 2 * BS_HALVES * 2)              // 251392 (aliases tiles 2-3; init only)
#define BIAS_OFF   (BS_OFF + BS_BYTES)                       // 302080
#define LDS_TOTAL  (BIAS_OFF + 1800 * 4)                     // 309280 (< 320 KB/WGP)

#define WS_BYTES_NEEDED ((size_t)(3 * HIDDEN) * HIDDEN * 2)  // 393216 (w_hh as bf16)

typedef __attribute__((ext_vector_type(16))) __bf16 v16bf;
typedef __attribute__((ext_vector_type(8)))  float  v8f;
typedef int v4i_ __attribute__((vector_size(16)));           // matches builtin param type

struct U4x2 { uint4 a, b; };

#if defined(__AMDGCN__) && __has_builtin(__builtin_amdgcn_global_load_async_to_lds_b128)
#define HAVE_ASYNC_LDS 1
#else
#define HAVE_ASYNC_LDS 0
#endif

#define GLOBAL_AS __attribute__((address_space(1)))
#define LDS_AS    __attribute__((address_space(3)))

// ---- bf16 truncation pack: one v_perm_b32 ----
__device__ __forceinline__ unsigned pack2bf(float a, float b) {
  return __builtin_amdgcn_perm(__builtin_bit_cast(unsigned, b),
                               __builtin_bit_cast(unsigned, a), 0x07060302u);
}
__device__ __forceinline__ unsigned short bf16of(float f) {
  return (unsigned short)(__builtin_bit_cast(unsigned, f) >> 16);
}

// A fragment (16x32 bf16): lane holds 16 contiguous K values k0..k0+15
__device__ __forceinline__ v16bf ldA(const unsigned short* p) {
  U4x2 t;
  t.a = *(const uint4*)p;
  t.b = *(const uint4*)(p + 8);
  return __builtin_bit_cast(v16bf, t);
}
// B fragment (32x16 bf16): lane holds 16 contiguous K values for one N column
__device__ __forceinline__ v16bf ldB(const unsigned short* p) {
  U4x2 t;
  t.a = *(const uint4*)p;
  t.b = *(const uint4*)(p + 8);
  return __builtin_bit_cast(v16bf, t);
}

// ---- fast gate activations (hardware v_tanh_f32) ----
#if defined(__AMDGCN__) && __has_builtin(__builtin_amdgcn_tanhf)
__device__ __forceinline__ float ftanh(float x) { return __builtin_amdgcn_tanhf(x); }
#else
__device__ __forceinline__ float ftanh(float x) {
  x = fminf(15.f, fmaxf(-15.f, x));
  float e = __builtin_amdgcn_exp2f(-2.8853900817779268f * x);   // exp(-2x)
  return (1.f - e) * __builtin_amdgcn_rcpf(1.f + e);
}
#endif
// sigmoid(x) with pre-halved bias bh: 0.5 + 0.5*tanh(0.5*x + bh)
__device__ __forceinline__ float fsigm_h(float x, float bh) {
  return fmaf(0.5f, ftanh(fmaf(0.5f, x, bh)), 0.5f);
}

template <int N>
__device__ __forceinline__ void wait_async_le() {
#if HAVE_ASYNC_LDS
#if __has_builtin(__builtin_amdgcn_s_wait_asynccnt)
  __builtin_amdgcn_s_wait_asynccnt(N);      // template arg is an ICE
#else
  asm volatile("s_wait_asynccnt %0" ::"i"(N) : "memory");
#endif
#endif
}

// stage TWO j-tiles of bf16 w_hh into a pair-buffer.
// threads 0-127 move tile jt0, threads 128-255 move tile jt0+1: 12 x 16B each,
// direct L2 -> LDS (ASYNCcnt), zero VGPR data traffic.
__device__ __forceinline__ void issue_pair(const unsigned short* __restrict__ whh_bf,
                                           unsigned short* bufBase, int pair, int tid) {
  int tile  = tid >> 7;              // 0 or 1
  int jt    = pair * 2 + tile;
  int cbase = (tid & 127) * 12;      // 1536 16-byte chunks per tile
  unsigned short* lbase = bufBase + tile * BS_HALVES;
#pragma unroll
  for (int i = 0; i < 12; ++i) {
    int c = cbase + i;
    int q = c * 8;                   // half index within tile
    int gate = q >> 12, n = (q >> 8) & 15, k = q & 255;
    const unsigned short* g = whh_bf + ((size_t)(gate * HIDDEN + jt * 16 + n)) * HIDDEN + k;
    unsigned short* l = lbase + (gate * 16 + n) * BS_STRIDE + k;
#if HAVE_ASYNC_LDS
    __builtin_amdgcn_global_load_async_to_lds_b128(
        (GLOBAL_AS v4i_*)g, (LDS_AS v4i_*)l, 0, 0);
#else
    *(uint4*)l = *(const uint4*)g;   // synchronous fallback
#endif
  }
}

// fp32-streaming fallback: fetch next single tile of fp32 w_hh into registers
__device__ __forceinline__ void prefetch_whh(const float* __restrict__ w_hh,
                                             int jt, int tid, float4* reg) {
#pragma unroll
  for (int i = 0; i < 12; ++i) {
    int c = tid * 12 + i;
    int q = c * 4;
    int gate = q >> 12, n = (q >> 8) & 15, k = q & 255;
    reg[i] = *(const float4*)(w_hh + (size_t)(gate * HIDDEN + jt * 16 + n) * HIDDEN + k);
  }
}

// one-time w_hh fp32 -> bf16 conversion into workspace
__global__ __launch_bounds__(THREADS)
void conv_whh_bf16(const float* __restrict__ w, unsigned short* __restrict__ o) {
  int idx = blockIdx.x * THREADS + threadIdx.x;  // 49152 float4 chunks
  int q = idx * 4;
  float4 v = *(const float4*)(w + q);
  uint2 p; p.x = pack2bf(v.x, v.y); p.y = pack2bf(v.z, v.w);
  *(uint2*)(o + q) = p;
}

template <bool USE_WS>
__global__ __launch_bounds__(THREADS)
void gru_gen_wmma(const float* __restrict__ z,
                  const float* __restrict__ fc_w,
                  const float* __restrict__ fc_b,
                  const float* __restrict__ b_ih,
                  const float* __restrict__ w_hh,
                  const float* __restrict__ b_hh,
                  const float* __restrict__ out_w,
                  const float* __restrict__ out_b,
                  const unsigned short* __restrict__ whh_bf,
                  float* __restrict__ outp)
{
  extern __shared__ char smem[];
  float*          hf    = (float*)smem;                        // h master (fp32)
  unsigned short* hbf   = (unsigned short*)(smem + HB_OFF);    // h as bf16 (A operand)
  unsigned short* Bsh   = (unsigned short*)(smem + BS_OFF);    // B tiles (4 slots)
  unsigned short* zbf   = (unsigned short*)(smem + ZB_OFF);    // z bf16 (aliases slots 2-3)
  float*          biasL = (float*)(smem + BIAS_OFF);

  const int tid  = threadIdx.x;
  const int wave = tid >> 5;
  const int lane = tid & 31;
  const int ln15 = lane & 15;
  const int lhi  = lane >> 4;
  const int r0   = blockIdx.x * ROWS_PER_BLOCK;

  // ---- biases: [0]=0.5*(bih_r+bhh_r), [256]=0.5*(bih_z+bhh_z),
  //      [512]=bih_n, [768]=bhh_n, [1024]=fc_b, [1280]=out_w(512), [1792]=out_b
  for (int i = tid; i < HIDDEN; i += THREADS) {
    biasL[i]          = 0.5f * (b_ih[i]          + b_hh[i]);
    biasL[256 + i]    = 0.5f * (b_ih[256 + i]    + b_hh[256 + i]);
    biasL[512 + i]    = b_ih[512 + i];
    biasL[768 + i]    = b_hh[512 + i];
    biasL[1024 + i]   = fc_b[i];
  }
  for (int i = tid; i < 512; i += THREADS) biasL[1280 + i] = out_w[i];
  if (tid < 2) biasL[1792 + tid] = out_b[tid];

  // ---- stage z rows (128 x 128 fp32 -> bf16) ----
#pragma unroll
  for (int i = 0; i < 16; ++i) {
    int c = tid + i * THREADS;
    int q = c * 4;
    int row = q >> 7, k = q & 127;
    float4 v = *(const float4*)(z + (size_t)(r0 + row) * LATENT + k);
    uint2 p; p.x = pack2bf(v.x, v.y); p.y = pack2bf(v.z, v.w);
    *(uint2*)(zbf + row * ZB_STRIDE + k) = p;
  }
  __syncthreads();

  // ---- h0 = tanh(z @ fc_w^T + fc_b), WMMA over K=128 (uses B slot 0) ----
  v16bf Za[4];
  {
    const unsigned short* ab = zbf + (wave * 16 + ln15) * ZB_STRIDE;
#pragma unroll
    for (int kk = 0; kk < 4; ++kk) Za[kk] = ldA(ab + kk * 32 + lhi * 16);
  }
  for (int jt = 0; jt < 16; ++jt) {
    __syncthreads();
#pragma unroll
    for (int i = 0; i < 2; ++i) {                 // stage fc_w j-tile: 16 x 128 fp32
      int c = tid * 2 + i;
      int q = c * 4;
      int n = q >> 7, k = q & 127;
      float4 v = *(const float4*)(fc_w + (size_t)(jt * 16 + n) * LATENT + k);
      uint2 p; p.x = pack2bf(v.x, v.y); p.y = pack2bf(v.z, v.w);
      *(uint2*)(Bsh + n * BS_STRIDE + k) = p;
    }
    __syncthreads();
    v8f acc = {};
    const unsigned short* bb = Bsh + ln15 * BS_STRIDE + lhi * 16;
#pragma unroll
    for (int kk = 0; kk < 4; ++kk) {
      v16bf bv = ldB(bb + kk * 32);
      acc = __builtin_amdgcn_wmma_f32_16x16x32_bf16(false, Za[kk], false, bv,
                                                    (short)0, acc, false, false);
    }
    int col  = jt * 16 + ln15;
    float fb = biasL[1024 + col];
    int rowb = wave * 16 + lhi * 8;
#pragma unroll
    for (int e = 0; e < 8; ++e) {
      float h0 = ftanh(acc[e] + fb);
      hf [(rowb + e) * HF_STRIDE + col] = h0;
      hbf[(rowb + e) * HB_STRIDE + col] = bf16of(h0);
    }
  }
  __syncthreads();   // B slots + zbf free for the recurrent streaming

  // ---- shared tile compute: 3-gate GEMM + GRU update for one j-tile ----
  v16bf Af[8];
  auto compute_tile = [&](const unsigned short* bsh, int jt) {
    v8f aR = {}, aZ = {}, aN = {};
    const unsigned short* bb = bsh + ln15 * BS_STRIDE + lhi * 16;
#pragma unroll
    for (int kk = 0; kk < 8; ++kk) {
      int kb = kk * 32;
      v16bf bR = ldB(bb + kb);
      v16bf bZ = ldB(bb + 16 * BS_STRIDE + kb);
      v16bf bN = ldB(bb + 32 * BS_STRIDE + kb);
      aR = __builtin_amdgcn_wmma_f32_16x16x32_bf16(false, Af[kk], false, bR, (short)0, aR, false, false);
      aZ = __builtin_amdgcn_wmma_f32_16x16x32_bf16(false, Af[kk], false, bZ, (short)0, aZ, false, false);
      aN = __builtin_amdgcn_wmma_f32_16x16x32_bf16(false, Af[kk], false, bN, (short)0, aN, false, false);
    }
    int col   = jt * 16 + ln15;
    float brh = biasL[col];            // 0.5*(b_ih_r + b_hh_r)
    float bzh = biasL[256 + col];      // 0.5*(b_ih_z + b_hh_z)
    float bin = biasL[512 + col];
    float bhn = biasL[768 + col];
    int rowb  = wave * 16 + lhi * 8;
#pragma unroll
    for (int e = 0; e < 8; ++e) {
      float r    = fsigm_h(aR[e], brh);
      float zg   = fsigm_h(aZ[e], bzh);
      float nn   = ftanh(fmaf(r, aN[e] + bhn, bin));
      float hold = hf[(rowb + e) * HF_STRIDE + col];
      float hnew = fmaf(zg, hold - nn, nn);              // (1-z)*n + z*h
      hf [(rowb + e) * HF_STRIDE + col] = hnew;
      hbf[(rowb + e) * HB_STRIDE + col] = bf16of(hnew);
    }
  };

  // ---- recurrent loop ----
  float4 wreg[12];
  if constexpr (USE_WS) issue_pair(whh_bf, Bsh, 0, tid);          // tiles 0,1 -> pair-buf 0
  else                  prefetch_whh(w_hh, 0, tid, wreg);

  for (int t = 0; t < SEQ_LEN; ++t) {
    // A fragments of current h (K = 256 -> 8 chunks), wave-private rows
    const unsigned short* ab = hbf + (wave * 16 + ln15) * HB_STRIDE;
#pragma unroll
    for (int kk = 0; kk < 8; ++kk) Af[kk] = ldA(ab + kk * 32 + lhi * 16);

    if constexpr (USE_WS) {
      for (int jp = 0; jp < 8; ++jp) {                   // 8 pairs of j-tiles
        __syncthreads();                                 // all done with other pair-buffer
        issue_pair(whh_bf, Bsh + ((jp + 1) & 1) * 2 * BS_HALVES, (jp + 1) & 7, tid);
        wait_async_le<12>();                             // my pair-jp transfers landed
        __syncthreads();                                 // pair jp visible block-wide
        const unsigned short* base = Bsh + (jp & 1) * 2 * BS_HALVES;
        compute_tile(base,             jp * 2);
        compute_tile(base + BS_HALVES, jp * 2 + 1);
      }
    } else {
      for (int jt = 0; jt < 16; ++jt) {
        __syncthreads();
#pragma unroll
        for (int i = 0; i < 12; ++i) {                   // commit prefetched fp32 tile
          int c = tid * 12 + i;
          int q = c * 4;
          int gate = q >> 12, n = (q >> 8) & 15, k = q & 255;
          uint2 p; p.x = pack2bf(wreg[i].x, wreg[i].y); p.y = pack2bf(wreg[i].z, wreg[i].w);
          *(uint2*)(Bsh + (gate * 16 + n) * BS_STRIDE + k) = p;
        }
        prefetch_whh(w_hh, (jt + 1) & 15, tid, wreg);
        __syncthreads();
        compute_tile(Bsh, jt);
      }
    }

    // fused output projection: out[b,t,c] = h . out_w[c] + out_b[c]
    {
      int row16 = ln15, ch = lhi;
      const float* hr = hf + (wave * 16 + row16) * HF_STRIDE;
      const float* wr = biasL + 1280 + ch * HIDDEN;
      float acc = biasL[1792 + ch];
#pragma unroll 8
      for (int jj = 0; jj < HIDDEN; jj += 4) {
        float4 h4 = *(const float4*)(hr + jj);
        float4 w4 = *(const float4*)(wr + jj);
        acc += h4.x * w4.x + h4.y * w4.y + h4.z * w4.z + h4.w * w4.w;
      }
      int gb = r0 + wave * 16 + row16;
      outp[((size_t)gb * SEQ_LEN + t) * 2 + ch] = acc;
    }
  }
}

extern "C" void kernel_launch(void* const* d_in, const int* in_sizes, int n_in,
                              void* d_out, int out_size, void* d_ws, size_t ws_size,
                              hipStream_t stream) {
  (void)in_sizes; (void)n_in; (void)out_size;
  const float* z     = (const float*)d_in[0];
  const float* fc_w  = (const float*)d_in[1];
  const float* fc_b  = (const float*)d_in[2];
  // d_in[3] = w_ih: unused (GRU input is all-zeros -> gi == b_ih)
  const float* b_ih  = (const float*)d_in[4];
  const float* w_hh  = (const float*)d_in[5];
  const float* b_hh  = (const float*)d_in[6];
  const float* out_w = (const float*)d_in[7];
  const float* out_b = (const float*)d_in[8];
  float* out = (float*)d_out;

  if (d_ws != nullptr && ws_size >= WS_BYTES_NEEDED) {
    unsigned short* whh_bf = (unsigned short*)d_ws;
    conv_whh_bf16<<<dim3((3 * HIDDEN * HIDDEN / 4) / THREADS), dim3(THREADS), 0, stream>>>(
        w_hh, whh_bf);
    gru_gen_wmma<true><<<dim3(NBLOCKS), dim3(THREADS), LDS_TOTAL, stream>>>(
        z, fc_w, fc_b, b_ih, w_hh, b_hh, out_w, out_b, whh_bf, out);
  } else {
    gru_gen_wmma<false><<<dim3(NBLOCKS), dim3(THREADS), LDS_TOTAL, stream>>>(
        z, fc_w, fc_b, b_ih, w_hh, b_hh, out_w, out_b, nullptr, out);
  }
}
